// BilateralNet_51410758533545
// MI455X (gfx1250) — matrix operand order repaired and checked by reading, not compiled, not run on hardware
//
#include <hip/hip_runtime.h>

typedef __attribute__((ext_vector_type(16))) _Float16 v16h;
typedef __attribute__((ext_vector_type(8)))  float    v8f;
typedef __fp16 pk2 __attribute__((ext_vector_type(2)));

#define NPIX    (2*160*160)
#define ROWS    164            // padded row stride (dwords): 36n mod 64 -> conflict-free b128
#define NFRAG   102            // 90 dense + 12 front-conv fragments

// ---------------------------------------------------------------------------
// helpers
// ---------------------------------------------------------------------------
__device__ __forceinline__ unsigned cvt_pk(float a, float b){   // v_cvt_pk_rtz_f16_f32
  union { pk2 h; unsigned u; } t;
  t.h = __builtin_amdgcn_cvt_pkrtz(a, b);
  return t.u;
}

__device__ __forceinline__ float relu(float a){                 // v_med3_f32
  return __builtin_amdgcn_fmed3f(a, 0.0f, __builtin_inff());
}
__device__ __forceinline__ float clampf(float a, float lo, float hi){
  return __builtin_amdgcn_fmed3f(a, lo, hi);
}

// read one tap of the (rotated, reflect-padded) input in original-frame coords
__device__ __forceinline__ float tap_load(const float* __restrict__ x, int base,
                                          int u, int v, int k){
  int ur = (u < 160) ? u : (318 - u);   // reflect (edge-excluded), p <= 2
  int vr = (v < 160) ? v : (318 - v);
  int io, jo;
  if      (k == 0){ io = ur;       jo = vr;       }
  else if (k == 1){ io = vr;       jo = 159 - ur; }
  else if (k == 2){ io = 159 - ur; jo = 159 - vr; }
  else            { io = 159 - vr; jo = ur;       }
  return x[base + io*160 + jo];
}

__device__ __forceinline__ float sigmoidf_(float v){ return 1.0f/(1.0f + __expf(-v)); }

// ---------------------------------------------------------------------------
// Prep: build all f16 WMMA A-fragments in the CDNA5 16-bit 16x32 A layout
// (lane = M row; VGPR v holds K pairs: lanes 0-15 -> K {0..7,16..23},
//  lanes 16-31 -> K {8..15,24..31}).
// Fragment table:
//   [0,8)   W2 (Kt=2)    [8,24)  W3 (Kt=4)   [24,48) W4 (Kt=6)
//   [48,80) W5 (Kt=8)    [80,90) W6 (Kt=10, rows>=4 zero)
//   [90,102) front conv: mode(x,s,c) x 4 M-tiles; K0..3 = taps, K4 = bias
// ---------------------------------------------------------------------------
__global__ void prep_frags(const float* __restrict__ w2, const float* __restrict__ w3,
                           const float* __restrict__ w4, const float* __restrict__ w5,
                           const float* __restrict__ w6,
                           const float* __restrict__ w1x, const float* __restrict__ b1x,
                           const float* __restrict__ w1s, const float* __restrict__ b1s,
                           const float* __restrict__ w1c, const float* __restrict__ b1c,
                           _Float16* __restrict__ frag){
  int tid = blockIdx.x*blockDim.x + threadIdx.x;
  if (tid >= NFRAG*32) return;
  int f = tid >> 5, lane = tid & 31;

  int mt, kt, K = 0, isLast = 0, isFront = 0;
  const float* W; const float* Bv = nullptr;
  if      (f <  8){ int lf = f;      mt = lf/2;  kt = lf%2;  K = 64;  W = w2; }
  else if (f < 24){ int lf = f - 8;  mt = lf/4;  kt = lf%4;  K = 128; W = w3; }
  else if (f < 48){ int lf = f - 24; mt = lf/6;  kt = lf%6;  K = 192; W = w4; }
  else if (f < 80){ int lf = f - 48; mt = lf/8;  kt = lf%8;  K = 256; W = w5; }
  else if (f < 90){ int lf = f - 80; mt = lf/10; kt = lf%10; K = 320; W = w6; isLast = 1; }
  else {
    int lf = f - 90; int mode = lf >> 2; mt = lf & 3; kt = 0; isFront = 1;
    if      (mode == 0){ W = w1x; Bv = b1x; }
    else if (mode == 1){ W = w1s; Bv = b1s; }
    else               { W = w1c; Bv = b1c; }
  }
  int m = mt*16 + (lane & 15);
  #pragma unroll
  for (int s = 0; s < 16; ++s){
    int v  = s >> 1, hh = s & 1;
    int kl = ((v < 4) ? 0 : 16) + ((lane >= 16) ? 8 : 0) + (v & 3)*2 + hh;
    float val = 0.f;
    if (isFront){
      if      (kl <  4) val = W[m*4 + kl];     // taps
      else if (kl == 4) val = Bv[m];           // bias column (B supplies 1.0)
    } else {
      int kg = kt*32 + kl;
      if (!isLast || m < 4) val = W[m*K + kg];
    }
    frag[(size_t)(f*32 + lane)*16 + s] = (_Float16)val;
  }
}

// ---------------------------------------------------------------------------
// Fused kernel. One wave per workgroup (barriers are S_NOP for single-wave
// workgroups); each wave owns 16 pixels. Activations in LDS transposed:
// sAct[pixel][K/2] pair-interleaved half2 dwords -> a lane's B fragment is
// 2 runs of 4 contiguous dwords, 4 apart (2 x ds_load_b128); layer outputs (8 consecutive
// channels per lane) pack to one ds_store_b128. Front conv is itself a WMMA
// (K=4 taps + bias column, zero-padded to 32).
// ---------------------------------------------------------------------------
__global__ __launch_bounds__(32, 1)
void bilateral_fused(const float* __restrict__ x,
                     const float* __restrict__ bb2, const float* __restrict__ bb3,
                     const float* __restrict__ bb4, const float* __restrict__ bb5,
                     const float* __restrict__ bb6,
                     const _Float16* __restrict__ fragw,
                     float* __restrict__ out)
{
  __shared__ float sB[260];                         // b2 b3 b4 b5 (64 ea) + b6 (4)
  __shared__ __align__(16) unsigned sAct[16][ROWS]; // [pixel][K/2] half2 dwords

  for (int t = threadIdx.x; t < 260; t += 32){
    float v;
    if      (t <  64) v = bb2[t];
    else if (t < 128) v = bb3[t-64];
    else if (t < 192) v = bb4[t-128];
    else if (t < 256) v = bb5[t-192];
    else              v = bb6[t-256];
    sB[t] = v;
  }
  __syncthreads();

  const int lane = threadIdx.x & 31;
  const int n    = lane & 15;
  const int hi   = lane >> 4;

  const int P    = blockIdx.x * 16 + n;             // 3200 blocks * 16 px
  const int bimg = P / 25600;
  const int rem  = P % 25600;
  const int i    = rem / 160;
  const int j    = rem % 160;
  const int base = bimg * 25600;

  const v16h* frags = (const v16h*)fragw;
  const v8f vz = {0.f,0.f,0.f,0.f,0.f,0.f,0.f,0.f};

  float par0 = 0.f, par1 = 0.f, par2 = 0.f, par3 = 0.f;

  for (int var = 0; var < 12; ++var){
    const int mode = var >> 2;                      // 0=x(2x2 d1) 1=s 2=c(2x2 d2)
    const int k    = var & 3;                       // rotation

    // position of this pixel in the rotated frame
    int h, w;
    if      (k == 0){ h = i;       w = j;       }
    else if (k == 1){ h = 159 - j; w = i;       }
    else if (k == 2){ h = 159 - i; w = 159 - j; }
    else            { h = j;       w = 159 - i; }

    int U0,V0,U1,V1,U2,V2,U3,V3;
    if (mode == 0){ U0=h;V0=w; U1=h;  V1=w+1; U2=h+1;V2=w;   U3=h+1;V3=w+1; }
    else if (mode == 1){ U0=h;V0=w; U1=h+1;V1=w+1; U2=h+1;V2=w+2; U3=h+2;V3=w+1; }
    else          { U0=h;V0=w; U1=h;  V1=w+2; U2=h+2;V2=w;   U3=h+2;V3=w+2; }

    const float t0 = tap_load(x, base, U0, V0, k);
    const float t1 = tap_load(x, base, U1, V1, k);
    const float t2 = tap_load(x, base, U2, V2, k);
    const float t3 = tap_load(x, base, U3, V3, k);

    // ---- front conv as WMMA: B = [taps, 1.0 (bias col), 0...] ------------
    union { v16h h; unsigned u[8]; } Bf;
    Bf.u[0] = hi ? 0u : cvt_pk(t0, t1);
    Bf.u[1] = hi ? 0u : cvt_pk(t2, t3);
    Bf.u[2] = hi ? 0u : cvt_pk(1.0f, 0.0f);
    Bf.u[3] = 0u; Bf.u[4] = 0u; Bf.u[5] = 0u; Bf.u[6] = 0u; Bf.u[7] = 0u;

    v8f fac[4];
    #pragma unroll
    for (int mt = 0; mt < 4; ++mt){
      v16h A = frags[(90 + mode*4 + mt)*32 + lane];
      fac[mt] = __builtin_amdgcn_wmma_f32_16x16x32_f16(
          false, A, false, Bf.h, (short)0, vz, false, false);
    }
    #pragma unroll
    for (int mt = 0; mt < 4; ++mt){
      unsigned pk[4];
      #pragma unroll
      for (int t = 0; t < 4; ++t)
        pk[t] = cvt_pk(relu(fac[mt][2*t]), relu(fac[mt][2*t+1]));
      *(uint4*)&sAct[n][mt*8 + hi*4] = make_uint4(pk[0], pk[1], pk[2], pk[3]);
    }
    __syncthreads();   // S_NOP barrier (single-wave WG) + LDS ordering

    const int KtA[4]   = {2, 4, 6, 8};
    const int fbase[4] = {0, 8, 24, 48};
    const int oOff[4]  = {64, 128, 192, 256};
    const int bOff[4]  = {0, 64, 128, 192};

    // dense block: D = W_L x act, M=64 (4 tiles), N=16 px, K = 64..256
    #pragma unroll
    for (int L = 0; L < 4; ++L){
      v8f acc[4]; acc[0]=vz; acc[1]=vz; acc[2]=vz; acc[3]=vz;
      const int Kt = KtA[L];
      for (int kt = 0; kt < Kt; ++kt){
        union { v16h h; uint4 q[2]; } Bm;                // B: 2 x 4 contiguous dwords
        Bm.q[0] = *(const uint4*)&sAct[n][kt*16 + hi*4];
        Bm.q[1] = *(const uint4*)&sAct[n][kt*16 + 8 + hi*4];
        #pragma unroll
        for (int mt = 0; mt < 4; ++mt){
          v16h A = frags[(fbase[L] + mt*Kt + kt)*32 + lane];
          acc[mt] = __builtin_amdgcn_wmma_f32_16x16x32_f16(
              false, A, false, Bm.h, (short)0, acc[mt], false, false);
        }
      }
      // bias + ReLU, append to concat: one b128 store per M-tile
      #pragma unroll
      for (int mt = 0; mt < 4; ++mt){
        const int bl = mt*16 + hi*8;
        unsigned pk[4];
        #pragma unroll
        for (int t = 0; t < 4; ++t){
          float a0 = relu(acc[mt][2*t]   + sB[bOff[L] + bl + 2*t]);
          float a1 = relu(acc[mt][2*t+1] + sB[bOff[L] + bl + 2*t + 1]);
          pk[t] = cvt_pk(a0, a1);
        }
        *(uint4*)&sAct[n][oOff[L]/2 + mt*8 + hi*4] = make_uint4(pk[0], pk[1], pk[2], pk[3]);
      }
      __syncthreads();
    }

    // final layer: 4 outputs, K = 320 (10 K-steps, single M tile)
    v8f accF = vz;
    #pragma unroll
    for (int kt = 0; kt < 10; ++kt){
      union { v16h h; uint4 q[2]; } Bm;
      Bm.q[0] = *(const uint4*)&sAct[n][kt*16 + hi*4];
      Bm.q[1] = *(const uint4*)&sAct[n][kt*16 + 8 + hi*4];
      v16h A = frags[(80 + kt)*32 + lane];
      accF = __builtin_amdgcn_wmma_f32_16x16x32_f16(
          false, A, false, Bm.h, (short)0, accF, false, false);
    }
    if (hi == 0){                 // rows 0..3 live in low half-wave, r=0..3
      par0 += accF[0] + sB[256];
      par1 += accF[1] + sB[257];
      par2 += accF[2] + sB[258];
      par3 += accF[3] + sB[259];
    }
    __syncthreads();
  }

  // ------------------- per-pixel epilogue (lanes 0..15) --------------------
  if (hi == 0){
    const float inv12 = 1.0f/12.0f;
    const float p0 = par0*inv12, p1 = par1*inv12, p2 = par2*inv12, p3 = par3*inv12;
    const float sigx = clampf(sigmoidf_(p0) + 1e-6f, 0.f, 1.f);
    const float sigy = clampf(sigmoidf_(p1) + 1e-6f, 0.f, 1.f);
    const float th   = clampf(tanhf(p2),          -1.f, 1.f);
    const float sr   = clampf(tanhf(p3) + 1e-6f,  -1.f, 1.f);

    out[NPIX   + P] = sigx;
    out[2*NPIX + P] = sigy;
    out[3*NPIX + P] = th;
    out[4*NPIX + P] = sr;

    const float sx = sigx*20.f, sy = sigy*20.f, srad = sr*10.f + 10.f;

    float patch[5][5];
    #pragma unroll
    for (int a = 0; a < 5; ++a)
      #pragma unroll
      for (int c = 0; c < 5; ++c){
        int xi = i + a - 2, xj = j + c - 2;
        bool inb = (xi >= 0) & (xi < 160) & (xj >= 0) & (xj < 160);
        patch[a][c] = inb ? x[base + xi*160 + xj] : 0.f;
      }

    const float inv2sr2 = 1.0f/(2.0f*srad*srad);
    float ssum = 0.f, val = 0.f;
    #pragma unroll
    for (int a = 0; a < 5; ++a)
      #pragma unroll
      for (int c = 0; c < 5; ++c){
        float tp  = patch[a][c];
        float dxv = fabsf(patch[2][c] - tp) * sx;
        float dyv = fabsf(patch[a][2] - tp) * sy;
        float sq  = (float)((a-2)*(a-2) + (c-2)*(c-2));
        float spat = __expf(-sq * inv2sr2);
        float col  = __expf(-0.5f*(dxv*dxv - 2.f*th*dxv*dyv + dyv*dyv));
        float kk = spat * col;
        ssum += kk;
        val  += kk * tp;
      }
    out[P] = val / ssum;
  }
}

// ---------------------------------------------------------------------------
extern "C" void kernel_launch(void* const* d_in, const int* in_sizes, int n_in,
                              void* d_out, int out_size, void* d_ws, size_t ws_size,
                              hipStream_t stream){
  (void)in_sizes; (void)n_in; (void)out_size; (void)ws_size;
  const float* x    = (const float*)d_in[0];
  const float* w1x  = (const float*)d_in[1];
  const float* b1x  = (const float*)d_in[2];
  const float* w1c  = (const float*)d_in[3];
  const float* b1c  = (const float*)d_in[4];
  const float* w1s  = (const float*)d_in[5];
  const float* b1s  = (const float*)d_in[6];
  const float* w2   = (const float*)d_in[7];
  const float* b2   = (const float*)d_in[8];
  const float* w3   = (const float*)d_in[9];
  const float* b3   = (const float*)d_in[10];
  const float* w4   = (const float*)d_in[11];
  const float* b4   = (const float*)d_in[12];
  const float* w5   = (const float*)d_in[13];
  const float* b5   = (const float*)d_in[14];
  const float* w6   = (const float*)d_in[15];
  const float* b6   = (const float*)d_in[16];

  _Float16* frag = (_Float16*)d_ws;   // 102 frags * 32 lanes * 32 B = 104448 B

  prep_frags<<<dim3(13), dim3(256), 0, stream>>>(
      w2, w3, w4, w5, w6, w1x, b1x, w1s, b1s, w1c, b1c, frag);
  bilateral_fused<<<dim3(3200), dim3(32), 0, stream>>>(
      x, b2, b3, b4, b5, b6, (const _Float16*)frag, (float*)d_out);
}
